// TemporalManifoldLearner_46961172414854
// MI455X (gfx1250) — matrix operand order repaired and checked by reading, not compiled, not run on hardware
//
#include <hip/hip_runtime.h>
#include <hip/hip_bf16.h>

typedef __attribute__((ext_vector_type(16))) __bf16 v16bf;
typedef __attribute__((ext_vector_type(8)))  float  v8f;

#define BATCH 64
#define SEQ   2048
#define DIN   768
#define DOUT  768
#define HM    16          // LSTM hidden
#define G4    (4*HM)      // 64 gates

__device__ __forceinline__ float sigmoidf_(float x) {
    return 1.0f / (1.0f + __expf(-x));
}

// ---------------------------------------------------------------------------
// Kernel 1: convert We1 [32,768] and We2 [16,32] to bf16 in workspace
// ---------------------------------------------------------------------------
__global__ void convert_weights(const float* __restrict__ We1,
                                const float* __restrict__ We2,
                                __bf16* __restrict__ We1bf,
                                __bf16* __restrict__ We2bf) {
    int idx = blockIdx.x * blockDim.x + threadIdx.x;
    const int n1 = 32 * DIN;
    const int n2 = 16 * 32;
    for (int i = idx; i < n1; i += gridDim.x * blockDim.x)
        We1bf[i] = (__bf16)We1[i];
    for (int i = idx; i < n2; i += gridDim.x * blockDim.x)
        We2bf[i] = (__bf16)We2[i];
}

// ---------------------------------------------------------------------------
// Kernel 2: emb = (relu(x @ We1.T + be1)) @ We2.T + be2   via WMMA bf16
// One wave handles a 16-row tile of the flattened [B*S, DIN] input.
// 8 waves per block -> grid = (B*S/16)/8 = 1024 blocks.
// ---------------------------------------------------------------------------
__global__ __launch_bounds__(256) void embed_wmma(
        const float* __restrict__ x,
        const __bf16* __restrict__ We1bf, const float* __restrict__ be1,
        const __bf16* __restrict__ We2bf, const float* __restrict__ be2,
        float* __restrict__ emb) {
    __shared__ __bf16 hstage[8][16][32];

    const int lane  = threadIdx.x & 31;
    const int wave  = threadIdx.x >> 5;
    const int laneN = lane & 15;     // A-row / B-col / C-col index
    const int half  = lane >> 4;     // K-half selector
    const long tile = (long)blockIdx.x * 8 + wave;   // 16-row tile id
    const long R    = tile * 16;                     // first row

    v8f c0 = {}; v8f c1 = {};

    // K loop over DIN in steps of 32 (bf16 WMMA K)
    for (int k0 = 0; k0 < DIN; k0 += 32) {
        // ---- A fragment: 16x32 bf16 from x (convert f32 -> bf16)
        // lane holds K = half*16 + 0..15 (elems 0..15)
        const float* xr = x + (R + laneN) * DIN + k0 + half * 16;
        float4 p0 = *(const float4*)(xr + 0);
        float4 p1 = *(const float4*)(xr + 4);
        float4 p2 = *(const float4*)(xr + 8);
        float4 p3 = *(const float4*)(xr + 12);
        v16bf a;
        a[0]=(__bf16)p0.x; a[1]=(__bf16)p0.y; a[2]=(__bf16)p0.z; a[3]=(__bf16)p0.w;
        a[4]=(__bf16)p1.x; a[5]=(__bf16)p1.y; a[6]=(__bf16)p1.z; a[7]=(__bf16)p1.w;
        a[8]=(__bf16)p2.x; a[9]=(__bf16)p2.y; a[10]=(__bf16)p2.z; a[11]=(__bf16)p2.w;
        a[12]=(__bf16)p3.x; a[13]=(__bf16)p3.y; a[14]=(__bf16)p3.z; a[15]=(__bf16)p3.w;

        // ---- B fragments: 32x16 bf16, B[k][n] = We1[n][k]
        // lane holds K = half*16 + 0..15 for column n = laneN (and laneN+16)
        v16bf b0 = *(const v16bf*)(We1bf + (size_t)laneN * DIN + k0 + half * 16);
        v16bf b1 = *(const v16bf*)(We1bf + (size_t)(16 + laneN) * DIN + k0 + half * 16);

        c0 = __builtin_amdgcn_wmma_f32_16x16x32_bf16(false, a, false, b0,
                                                     (short)0, c0, false, false);
        c1 = __builtin_amdgcn_wmma_f32_16x16x32_bf16(false, a, false, b1,
                                                     (short)0, c1, false, false);
    }

    // bias + relu, stage h (16 rows x 32 feats) to LDS in bf16 for 2nd GEMM.
    // C layout: c[r] = element (row = r + 8*half, col = laneN)
    float bia0 = be1[laneN];
    float bia1 = be1[16 + laneN];
    #pragma unroll
    for (int r = 0; r < 8; ++r) {
        float h0 = fmaxf(c0[r] + bia0, 0.0f);
        float h1 = fmaxf(c1[r] + bia1, 0.0f);
        hstage[wave][r + 8 * half][laneN]      = (__bf16)h0;
        hstage[wave][r + 8 * half][16 + laneN] = (__bf16)h1;
    }
    __syncthreads();

    // ---- second GEMM: emb_tile[16,16] = h[16,32] @ We2.T[32,16]
    v16bf a2;
    #pragma unroll
    for (int j = 0; j < 8; ++j) {
        a2[j]     = hstage[wave][laneN][half * 16 + j];
        a2[8 + j] = hstage[wave][laneN][half * 16 + 8 + j];
    }
    v16bf b2 = *(const v16bf*)(We2bf + laneN * 32 + half * 16);
    v8f z = {};
    v8f c2 = __builtin_amdgcn_wmma_f32_16x16x32_bf16(false, a2, false, b2,
                                                     (short)0, z, false, false);
    float b2bias = be2[laneN];
    #pragma unroll
    for (int r = 0; r < 8; ++r)
        emb[(R + r + 8 * half) * HM + laneN] = c2[r] + b2bias;
}

// ---------------------------------------------------------------------------
// Kernel 3: fused 2-layer LSTM scan. One wave32 per batch element.
// Lane l owns gates (2l, 2l+1) of both layers; lanes 0..15 own h/c elements.
// State + all 4 weight matrices live in registers; cross-lane via __shfl.
// ---------------------------------------------------------------------------
__global__ __launch_bounds__(32, 1) void lstm_scan(
        const float* __restrict__ emb,
        const float* __restrict__ Wih0, const float* __restrict__ Whh0,
        const float* __restrict__ bih0, const float* __restrict__ bhh0,
        const float* __restrict__ Wih1, const float* __restrict__ Whh1,
        const float* __restrict__ bih1, const float* __restrict__ bhh1,
        float* __restrict__ flow) {
    const int b = blockIdx.x;
    const int l = threadIdx.x;
    const int ga = 2 * l, gb = 2 * l + 1;

    float wih0a[HM], wih0b[HM], whh0a[HM], whh0b[HM];
    float wih1a[HM], wih1b[HM], whh1a[HM], whh1b[HM];
    #pragma unroll
    for (int k = 0; k < HM; k += 4) {
        float4 t;
        t = *(const float4*)(Wih0 + ga*HM + k); wih0a[k]=t.x; wih0a[k+1]=t.y; wih0a[k+2]=t.z; wih0a[k+3]=t.w;
        t = *(const float4*)(Wih0 + gb*HM + k); wih0b[k]=t.x; wih0b[k+1]=t.y; wih0b[k+2]=t.z; wih0b[k+3]=t.w;
        t = *(const float4*)(Whh0 + ga*HM + k); whh0a[k]=t.x; whh0a[k+1]=t.y; whh0a[k+2]=t.z; whh0a[k+3]=t.w;
        t = *(const float4*)(Whh0 + gb*HM + k); whh0b[k]=t.x; whh0b[k+1]=t.y; whh0b[k+2]=t.z; whh0b[k+3]=t.w;
        t = *(const float4*)(Wih1 + ga*HM + k); wih1a[k]=t.x; wih1a[k+1]=t.y; wih1a[k+2]=t.z; wih1a[k+3]=t.w;
        t = *(const float4*)(Wih1 + gb*HM + k); wih1b[k]=t.x; wih1b[k+1]=t.y; wih1b[k+2]=t.z; wih1b[k+3]=t.w;
        t = *(const float4*)(Whh1 + ga*HM + k); whh1a[k]=t.x; whh1a[k+1]=t.y; whh1a[k+2]=t.z; whh1a[k+3]=t.w;
        t = *(const float4*)(Whh1 + gb*HM + k); whh1b[k]=t.x; whh1b[k+1]=t.y; whh1b[k+2]=t.z; whh1b[k+3]=t.w;
    }
    const float bias0a = bih0[ga] + bhh0[ga], bias0b = bih0[gb] + bhh0[gb];
    const float bias1a = bih1[ga] + bhh1[ga], bias1b = bih1[gb] + bhh1[gb];

    float h0v = 0.0f, c0v = 0.0f, h1v = 0.0f, c1v = 0.0f;

    const int jh = l >> 1, js = l & 1;        // gate-gather source for lane l (=j)
    const float* embrow = emb + ((size_t)b * SEQ) * HM;
    float* flowrow = flow + ((size_t)b * SEQ) * HM;

    for (int t = 0; t < SEQ; ++t) {
        // ---- layer 0: g = emb_t @ Wih0.T + h0 @ Whh0.T + bias
        float ev = embrow[(size_t)t * HM + (l & 15)];
        float a0a = bias0a, a0b = bias0b;
        #pragma unroll
        for (int k = 0; k < HM; ++k) {
            float ek = __shfl(ev, k);
            float hk = __shfl(h0v, k);
            a0a += ek * wih0a[k] + hk * whh0a[k];
            a0b += ek * wih0b[k] + hk * whh0b[k];
        }
        // gather (i,f,g,o)[j=l] : gate j, 16+j, 32+j, 48+j
        float t0, t1;
        t0 = __shfl(a0a, jh);      t1 = __shfl(a0b, jh);      float gi = js ? t1 : t0;
        t0 = __shfl(a0a, 8 + jh);  t1 = __shfl(a0b, 8 + jh);  float gf = js ? t1 : t0;
        t0 = __shfl(a0a, 16 + jh); t1 = __shfl(a0b, 16 + jh); float gg = js ? t1 : t0;
        t0 = __shfl(a0a, 24 + jh); t1 = __shfl(a0b, 24 + jh); float go = js ? t1 : t0;
        {
            float ii = sigmoidf_(gi), ff = sigmoidf_(gf);
            float g2 = tanhf(gg),     oo = sigmoidf_(go);
            c0v = ff * c0v + ii * g2;
            h0v = oo * tanhf(c0v);          // valid in lanes 0..15
        }
        // ---- layer 1: g = h0 @ Wih1.T + h1 @ Whh1.T + bias
        float a1a = bias1a, a1b = bias1b;
        #pragma unroll
        for (int k = 0; k < HM; ++k) {
            float fk = __shfl(h0v, k);
            float hk = __shfl(h1v, k);
            a1a += fk * wih1a[k] + hk * whh1a[k];
            a1b += fk * wih1b[k] + hk * whh1b[k];
        }
        t0 = __shfl(a1a, jh);      t1 = __shfl(a1b, jh);      gi = js ? t1 : t0;
        t0 = __shfl(a1a, 8 + jh);  t1 = __shfl(a1b, 8 + jh);  gf = js ? t1 : t0;
        t0 = __shfl(a1a, 16 + jh); t1 = __shfl(a1b, 16 + jh); gg = js ? t1 : t0;
        t0 = __shfl(a1a, 24 + jh); t1 = __shfl(a1b, 24 + jh); go = js ? t1 : t0;
        {
            float ii = sigmoidf_(gi), ff = sigmoidf_(gf);
            float g2 = tanhf(gg),     oo = sigmoidf_(go);
            c1v = ff * c1v + ii * g2;
            h1v = oo * tanhf(c1v);
        }
        if (l < HM) flowrow[(size_t)t * HM + l] = h1v;
    }
}

// ---------------------------------------------------------------------------
// Kernel 4: curvature + consistency + weighted temporal mean. One WG / batch.
// Deterministic serial per-feature reduction over LDS partials (no float atomics).
// ---------------------------------------------------------------------------
__global__ __launch_bounds__(256) void temporal_reduce(
        const float* __restrict__ emb, const float* __restrict__ flow,
        const float* __restrict__ Wc1, const float* __restrict__ bc1,
        const float* __restrict__ Wc2, const float* __restrict__ bc2,
        float* __restrict__ temporal) {
    __shared__ float sWc1[16][32];
    __shared__ float sbc1[16], sWc2[16];
    __shared__ float sbc2;
    __shared__ float rbuf[256 * HM];

    const int tid = threadIdx.x;
    const int b   = blockIdx.x;
    const float eps = 1e-8f;

    if (tid < 16) { sbc1[tid] = bc1[tid]; sWc2[tid] = Wc2[tid]; }
    if (tid == 0) sbc2 = bc2[0];
    for (int i = tid; i < 512; i += 256) sWc1[i >> 5][i & 31] = Wc1[i];
    __syncthreads();

    float acc[HM];
    #pragma unroll
    for (int j = 0; j < HM; ++j) acc[j] = 0.0f;

    const float* er = emb  + (size_t)b * SEQ * HM;
    const float* fr = flow + (size_t)b * SEQ * HM;

    for (int s = tid; s < SEQ; s += 256) {
        // curvature
        float curv = 0.0f;
        if (s > 0 && s < SEQ - 1) {
            float fsum = 0.0f, ssum = 0.0f;
            #pragma unroll
            for (int j = 0; j < HM; ++j) {
                float p = er[(size_t)(s - 1) * HM + j];
                float c = er[(size_t)s * HM + j];
                float n = er[(size_t)(s + 1) * HM + j];
                float d1 = c - p;
                float d2 = n - 2.0f * c + p;
                fsum += d1 * d1; ssum += d2 * d2;
            }
            curv = sqrtf(ssum) / (sqrtf(fsum) + eps);
        }
        // consistency (clamped index, matching reference edge-replication)
        int sc = (s < SEQ - 1) ? s : SEQ - 2;
        float f0[HM], f1[HM], fs[HM];
        #pragma unroll
        for (int j = 0; j < HM; ++j) {
            f0[j] = fr[(size_t)sc * HM + j];
            f1[j] = fr[(size_t)(sc + 1) * HM + j];
            fs[j] = fr[(size_t)s * HM + j];
        }
        float z = sbc2;
        #pragma unroll
        for (int j = 0; j < 16; ++j) {
            float hc = sbc1[j];
            #pragma unroll
            for (int k = 0; k < HM; ++k) hc += sWc1[j][k] * f0[k];
            #pragma unroll
            for (int k = 0; k < HM; ++k) hc += sWc1[j][16 + k] * f1[k];
            z += fmaxf(hc, 0.0f) * sWc2[j];
        }
        float cons = sigmoidf_(z);
        float w = cons / (curv + eps);
        #pragma unroll
        for (int j = 0; j < HM; ++j) acc[j] += fs[j] * w;
    }

    #pragma unroll
    for (int j = 0; j < HM; ++j) rbuf[tid * HM + j] = acc[j];
    __syncthreads();
    if (tid < HM) {
        float s = 0.0f;
        for (int t = 0; t < 256; ++t) s += rbuf[t * HM + tid];
        temporal[b * HM + tid] = s * (1.0f / (float)SEQ);
    }
}

// ---------------------------------------------------------------------------
// Kernel 5/6: head MLP (tiny; fp32 VALU for accuracy)
// ---------------------------------------------------------------------------
__global__ void head1(const float* __restrict__ temporal,
                      const float* __restrict__ Wd1, const float* __restrict__ bd1,
                      float* __restrict__ hidden) {
    int idx = blockIdx.x * blockDim.x + threadIdx.x;   // 64*1536
    if (idx >= BATCH * 2 * DOUT) return;
    int b = idx / (2 * DOUT), o = idx % (2 * DOUT);
    float s = bd1[o];
    #pragma unroll
    for (int k = 0; k < HM; ++k) s += Wd1[o * HM + k] * temporal[b * HM + k];
    hidden[idx] = fmaxf(s, 0.0f);
}

__global__ void head2(const float* __restrict__ hidden,
                      const float* __restrict__ Wd2, const float* __restrict__ bd2,
                      float* __restrict__ out) {
    int idx = blockIdx.x * blockDim.x + threadIdx.x;   // 64*768
    if (idx >= BATCH * DOUT) return;
    int b = idx / DOUT, o = idx % DOUT;
    const float* hr = hidden + (size_t)b * 2 * DOUT;
    const float* wr = Wd2 + (size_t)o * 2 * DOUT;
    float s = bd2[o];
    for (int k = 0; k < 2 * DOUT; k += 4) {
        float4 h = *(const float4*)(hr + k);
        float4 w = *(const float4*)(wr + k);
        s += h.x * w.x + h.y * w.y + h.z * w.z + h.w * w.w;
    }
    out[idx] = s;
}

// ---------------------------------------------------------------------------
extern "C" void kernel_launch(void* const* d_in, const int* in_sizes, int n_in,
                              void* d_out, int out_size, void* d_ws, size_t ws_size,
                              hipStream_t stream) {
    (void)in_sizes; (void)n_in; (void)out_size; (void)ws_size;
    const float* x    = (const float*)d_in[0];
    const float* We1  = (const float*)d_in[1];
    const float* be1  = (const float*)d_in[2];
    const float* We2  = (const float*)d_in[3];
    const float* be2  = (const float*)d_in[4];
    const float* Wih0 = (const float*)d_in[5];
    const float* Whh0 = (const float*)d_in[6];
    const float* bih0 = (const float*)d_in[7];
    const float* bhh0 = (const float*)d_in[8];
    const float* Wih1 = (const float*)d_in[9];
    const float* Whh1 = (const float*)d_in[10];
    const float* bih1 = (const float*)d_in[11];
    const float* bhh1 = (const float*)d_in[12];
    const float* Wc1  = (const float*)d_in[13];
    const float* bc1  = (const float*)d_in[14];
    const float* Wc2  = (const float*)d_in[15];
    const float* bc2  = (const float*)d_in[16];
    const float* Wd1  = (const float*)d_in[17];
    const float* bd1  = (const float*)d_in[18];
    const float* Wd2  = (const float*)d_in[19];
    const float* bd2  = (const float*)d_in[20];
    float* out = (float*)d_out;

    // workspace layout
    char* ws = (char*)d_ws;
    __bf16* We1bf   = (__bf16*)ws;                              // 48 KB
    __bf16* We2bf   = (__bf16*)(ws + 49152);                    // 1 KB
    float* emb      = (float*)(ws + 65536);                     // 8 MB
    float* flow     = (float*)(ws + 65536 + 8388608);           // 8 MB
    float* temporal = (float*)(ws + 65536 + 2 * 8388608);       // 4 KB
    float* hidden   = (float*)(ws + 65536 + 2 * 8388608 + 4096);// 384 KB

    convert_weights<<<64, 256, 0, stream>>>(We1, We2, We1bf, We2bf);
    embed_wmma<<<(BATCH * SEQ / 16) / 8, 256, 0, stream>>>(x, We1bf, be1, We2bf, be2, emb);
    lstm_scan<<<BATCH, 32, 0, stream>>>(emb, Wih0, Whh0, bih0, bhh0,
                                        Wih1, Whh1, bih1, bhh1, flow);
    temporal_reduce<<<BATCH, 256, 0, stream>>>(emb, flow, Wc1, bc1, Wc2, bc2, temporal);
    head1<<<(BATCH * 2 * DOUT + 255) / 256, 256, 0, stream>>>(temporal, Wd1, bd1, hidden);
    head2<<<(BATCH * DOUT + 255) / 256, 256, 0, stream>>>(hidden, Wd2, bd2, out);
}
